// FeedBack_56642028700237
// MI455X (gfx1250) — compile-verified, matrix operand order checked by reading, not run on hardware
//
#include <hip/hip_runtime.h>
#include <hip/hip_bf16.h>

typedef __bf16 bf16;
typedef __attribute__((ext_vector_type(16))) __bf16 v16bf;
typedef __attribute__((ext_vector_type(8)))  __bf16 v8bf;
typedef __attribute__((ext_vector_type(8)))  float  v8f;

#define Bsz    1024
#define F_IN   32
#define UNITS  512
#define T_IN   288
#define T_OUT  288
#define KTOT   (UNITS + F_IN)     // 544 (= 1088 bytes per WcT row, 16B multiple)
#define GCOLS  (4 * UNITS)        // 2048

#define BM 64      // batch tile
#define BU 64      // unit tile
#define BK 32      // K chunk (matches wmma K)
#define NKC 17     // 544 / 32
#define LDA 40     // tileA stride (bf16): 80B = 20 banks -> conflict-free, 16B aligned
#define LDBT 40    // tileBt stride (bf16): fragment-major B tile [col][K]

// ---------------------------------------------------------------------------
// Build transposed bf16 weights WcT [2048 cols][544 K] from fp32 R[512,2048],
// W[32,2048].  WcT[col][k] = (k<512 ? R[k][col] : W[k-512][col])
// ---------------------------------------------------------------------------
__global__ __launch_bounds__(256)
void convert_wc_kernel(const float* __restrict__ R, const float* __restrict__ W,
                       bf16* __restrict__ WcT) {
    int idx = blockIdx.x * 256 + threadIdx.x;      // over 544*2048
    if (idx >= KTOT * GCOLS) return;
    int k   = idx >> 11;          // /2048
    int col = idx & (GCOLS - 1);
    float v = (k < UNITS) ? R[k * GCOLS + col] : W[(k - UNITS) * GCOLS + col];
    WcT[(size_t)col * KTOT + k] = (bf16)v;
}

// ---------------------------------------------------------------------------
// One LSTM timestep:  z = [h|x] @ [R;W] + b ; gate math ; update c, h_out
// grid = (1024/64, 512/64), block = 256 (8 wave32)
// Double-buffered LDS: global staging of chunk kc+1 overlaps WMMA on chunk kc.
// ---------------------------------------------------------------------------
__global__ __launch_bounds__(256)
void lstm_step_kernel(const bf16*  __restrict__ WcT,    // [2048][544] bf16
                      const float* __restrict__ bias,   // [2048]
                      const float* __restrict__ xsrc,   // x rows, stride xstride
                      int xstride,
                      const float* __restrict__ hin,    // [1024][512]
                      float*       __restrict__ hout,   // [1024][512]
                      float*       __restrict__ cst)    // [1024][512] in/out
{
    __shared__ bf16 tileA [2][BM * LDA];    // 64 rows x 32 K (row-major, padded)
    __shared__ bf16 tileBt[2][256 * LDBT];  // 256 cols (4 gates x 64 u) x 32 K

    const int tid    = threadIdx.x;
    const int lane   = tid & 31;
    const int wave   = tid >> 5;
    const int blockM = blockIdx.x * BM;
    const int blockU = blockIdx.y * BU;

    // wave -> 16-row strip, two 16-unit patches, all 4 gates
    const int mstrip = (wave & 3) * 16;
    const int u0     = (wave >> 2) * 32;

    const int nl = lane & 15;   // N (or M) index within fragment
    const int h2 = lane >> 4;   // half-wave selector

    // staging maps
    const int arow   = tid >> 2;          // 0..63
    const int akbase = (tid & 3) * 8;     // 0,8,16,24
    const int bgate  = tid >> 6;          // 0..3
    const int bu     = tid & 63;          // 0..63
    const size_t brow = (size_t)(bgate * UNITS + blockU + bu) * KTOT; // WcT row

    v8f acc[2][4];
    {
        v8f z = {};
        #pragma unroll
        for (int p = 0; p < 2; ++p)
            #pragma unroll
            for (int g = 0; g < 4; ++g) acc[p][g] = z;
    }

    // ---- prologue: stage chunk 0 into buffer 0 ----
    {
        const float* srow = hin + (size_t)(blockM + arow) * UNITS + akbase;
        float4 a0 = ((const float4*)srow)[0];
        float4 a1 = ((const float4*)srow)[1];
        v8bf av;
        av[0] = (bf16)a0.x; av[1] = (bf16)a0.y;
        av[2] = (bf16)a0.z; av[3] = (bf16)a0.w;
        av[4] = (bf16)a1.x; av[5] = (bf16)a1.y;
        av[6] = (bf16)a1.z; av[7] = (bf16)a1.w;
        *(v8bf*)&tileA[0][arow * LDA + akbase] = av;

        const bf16* src = WcT + brow;
        bf16*       dst = &tileBt[0][tid * LDBT];
        #pragma unroll
        for (int j = 0; j < 4; ++j)
            *(v8bf*)(dst + 8 * j) = *(const v8bf*)(src + 8 * j);
    }
    __syncthreads();

    for (int kc = 0; kc < NKC; ++kc) {
        const int cur = kc & 1;
        const int nxt = cur ^ 1;
        const int cn  = kc + 1;                 // chunk being staged
        const bool have_next = (cn < NKC);

        // ---- issue next chunk's global loads into registers (overlaps WMMA) ----
        float4 a0, a1;
        v8bf bblk[4];
        if (have_next) {
            const float* srow = (cn < 16)
                ? (hin  + (size_t)(blockM + arow) * UNITS + cn * BK + akbase)
                : (xsrc + (size_t)(blockM + arow) * xstride + akbase);
            a0 = ((const float4*)srow)[0];
            a1 = ((const float4*)srow)[1];
            const bf16* src = WcT + brow + cn * BK;     // 64B contiguous
            #pragma unroll
            for (int j = 0; j < 4; ++j)
                bblk[j] = ((const v8bf*)src)[j];
            __builtin_prefetch(src + BK, 0, 1);         // chunk cn+1 (contiguous)
        }

        // ---- A fragment: two contiguous 16B runs -> 2x ds_load_b128 ----
        const bf16* arp = &tileA[cur][(mstrip + nl) * LDA + h2 * 8];
        v8bf alo = *(const v8bf*)(arp);
        v8bf ahi = *(const v8bf*)(arp + 16);
        v16bf afrag = __builtin_shufflevector(alo, ahi,
            0, 1, 2, 3, 4, 5, 6, 7, 8, 9, 10, 11, 12, 13, 14, 15);

        // ---- 8 WMMAs: 2 unit patches x 4 gates ----
        #pragma unroll
        for (int p = 0; p < 2; ++p) {
            #pragma unroll
            for (int g = 0; g < 4; ++g) {
                const int colb = g * 64 + u0 + p * 16 + nl;
                const bf16* brp = &tileBt[cur][colb * LDBT + h2 * 8];
                v8bf blo = *(const v8bf*)(brp);
                v8bf bhi = *(const v8bf*)(brp + 16);
                v16bf bfrag = __builtin_shufflevector(blo, bhi,
                    0, 1, 2, 3, 4, 5, 6, 7, 8, 9, 10, 11, 12, 13, 14, 15);
                acc[p][g] = __builtin_amdgcn_wmma_f32_16x16x32_bf16(
                    false, afrag, false, bfrag, (short)0, acc[p][g], false, false);
            }
        }

        // ---- commit staged chunk to the other LDS buffer ----
        if (have_next) {
            v8bf av;
            av[0] = (bf16)a0.x; av[1] = (bf16)a0.y;
            av[2] = (bf16)a0.z; av[3] = (bf16)a0.w;
            av[4] = (bf16)a1.x; av[5] = (bf16)a1.y;
            av[6] = (bf16)a1.z; av[7] = (bf16)a1.w;
            *(v8bf*)&tileA[nxt][arow * LDA + akbase] = av;
            bf16* dst = &tileBt[nxt][tid * LDBT];
            #pragma unroll
            for (int j = 0; j < 4; ++j)
                *(v8bf*)(dst + 8 * j) = bblk[j];
        }
        __syncthreads();   // single barrier per chunk
    }

    // ---- epilogue: gates in registers (C layout: M = v + 8*h2, N = nl) ----
    #pragma unroll
    for (int p = 0; p < 2; ++p) {
        const int col = blockU + u0 + p * 16 + nl;        // unit 0..511
        const float bi  = bias[0 * UNITS + col];
        const float bf_ = bias[1 * UNITS + col];
        const float bg  = bias[2 * UNITS + col];
        const float bo  = bias[3 * UNITS + col];
        #pragma unroll
        for (int v = 0; v < 8; ++v) {
            const int row  = blockM + mstrip + v + 8 * h2;
            const size_t s = (size_t)row * UNITS + col;
            float iv = 1.0f / (1.0f + __expf(-(acc[p][0][v] + bi)));
            float fv = 1.0f / (1.0f + __expf(-(acc[p][1][v] + bf_)));
            float gv = tanhf(acc[p][2][v] + bg);
            float ov = 1.0f / (1.0f + __expf(-(acc[p][3][v] + bo)));
            float cn = fv * cst[s] + iv * gv;
            cst[s]  = cn;
            hout[s] = ov * tanhf(cn);
        }
    }
}

// ---------------------------------------------------------------------------
// Projection: pred = h @ Wd + bd ; writes d_out slice and feedback buffer
// ---------------------------------------------------------------------------
__global__ __launch_bounds__(256)
void proj_kernel(const float* __restrict__ h, const float* __restrict__ Wd,
                 const float* __restrict__ bd, float* __restrict__ out_t,
                 float* __restrict__ predbuf) {
    int idx = blockIdx.x * 256 + threadIdx.x;   // 1024*32
    int b = idx >> 5;
    int f = idx & 31;
    float acc = bd[f];
    const float* hr = h + (size_t)b * UNITS;
    #pragma unroll 8
    for (int u = 0; u < UNITS; ++u)
        acc = fmaf(hr[u], Wd[u * F_IN + f], acc);
    out_t[(size_t)b * (T_OUT * F_IN) + f] = acc;
    predbuf[idx] = acc;
}

// ---------------------------------------------------------------------------
extern "C" void kernel_launch(void* const* d_in, const int* in_sizes, int n_in,
                              void* d_out, int out_size, void* d_ws, size_t ws_size,
                              hipStream_t stream) {
    const float* inputs = (const float*)d_in[0];  // [1024,288,32]
    const float* W      = (const float*)d_in[1];  // [32,2048]
    const float* R      = (const float*)d_in[2];  // [512,2048]
    const float* b      = (const float*)d_in[3];  // [2048]
    const float* Wd     = (const float*)d_in[4];  // [512,32]
    const float* bd     = (const float*)d_in[5];  // [32]
    float* out = (float*)d_out;                   // [1024,288,32]

    // workspace layout: WcT(bf16) | h0 | c | h1 | pred
    char* ws = (char*)d_ws;
    size_t off = 0;
    bf16*  WcT = (bf16*)(ws + off); off += (size_t)GCOLS * KTOT * sizeof(bf16);   // 2228224
    off = (off + 255) & ~(size_t)255;
    float* h0 = (float*)(ws + off); off += (size_t)Bsz * UNITS * sizeof(float);
    float* c  = (float*)(ws + off); off += (size_t)Bsz * UNITS * sizeof(float);
    float* h1 = (float*)(ws + off); off += (size_t)Bsz * UNITS * sizeof(float);
    float* pr = (float*)(ws + off); off += (size_t)Bsz * F_IN * sizeof(float);

    // one-time per launch: weight convert/transpose + zero h0,c (contiguous)
    convert_wc_kernel<<<(KTOT * GCOLS + 255) / 256, 256, 0, stream>>>(R, W, WcT);
    hipMemsetAsync(h0, 0, (size_t)2 * Bsz * UNITS * sizeof(float), stream);

    dim3 sgrid(Bsz / BM, UNITS / BU);   // 16 x 8
    float* hin = h0;
    float* hout = h1;

    // warmup over the 288 input timesteps
    for (int t = 0; t < T_IN; ++t) {
        lstm_step_kernel<<<sgrid, 256, 0, stream>>>(
            WcT, b, inputs + (size_t)t * F_IN, T_IN * F_IN, hin, hout, c);
        float* tmp = hin; hin = hout; hout = tmp;
    }
    // first prediction
    proj_kernel<<<(Bsz * F_IN) / 256, 256, 0, stream>>>(hin, Wd, bd, out + 0 * F_IN, pr);
    // autoregressive decode
    for (int s = 1; s < T_OUT; ++s) {
        lstm_step_kernel<<<sgrid, 256, 0, stream>>>(WcT, b, pr, F_IN, hin, hout, c);
        float* tmp = hin; hin = hout; hout = tmp;
        proj_kernel<<<(Bsz * F_IN) / 256, 256, 0, stream>>>(hin, Wd, bd, out + (size_t)s * F_IN, pr);
    }
}
